// ImprovedGraphSAGE_74053826117923
// MI455X (gfx1250) — matrix operand order repaired and checked by reading, not compiled, not run on hardware
//
#include <hip/hip_runtime.h>
#include <math.h>

typedef __attribute__((ext_vector_type(16))) _Float16 v16h;
typedef __attribute__((ext_vector_type(8)))  float    v8f;

#define HID 128

// ---------------- degree / norm prep ----------------

__global__ void k_deg_init(float* __restrict__ deg, int n) {
  int i = blockIdx.x * blockDim.x + threadIdx.x;
  if (i < n) deg[i] = 1.0f;   // self loop
}

__global__ void k_deg_count(const long long* __restrict__ col, long long E,
                            float* __restrict__ deg) {
  long long e = (long long)blockIdx.x * blockDim.x + threadIdx.x;
  if (e < E) atomicAdd(&deg[(int)col[e]], 1.0f);
}

__global__ void k_dinv(const float* __restrict__ deg, float* __restrict__ dinv,
                       float* __restrict__ invdeg, int n) {
  int i = blockIdx.x * blockDim.x + threadIdx.x;
  if (i < n) {
    float d = deg[i];
    dinv[i]   = rsqrtf(d);
    invdeg[i] = 1.0f / d;
  }
}

__global__ void k_norm(const long long* __restrict__ ei, long long E,
                       const float* __restrict__ dinv, float* __restrict__ norm) {
  long long e = (long long)blockIdx.x * blockDim.x + threadIdx.x;
  if (e < E) norm[e] = dinv[(int)ei[e]] * dinv[(int)ei[E + e]];
}

// ---------------- dtype conversion ----------------

__global__ void k_cvt_f16(const float* __restrict__ src, _Float16* __restrict__ dst, int n) {
  int i = blockIdx.x * blockDim.x + threadIdx.x;
  if (i < n) dst[i] = (_Float16)src[i];
}

__global__ void k_premul(const float* __restrict__ x, const float* __restrict__ imp,
                         _Float16* __restrict__ a16, long long n) {
  long long i = (long long)blockIdx.x * blockDim.x + threadIdx.x;
  if (i < n) a16[i] = (_Float16)(x[i] * imp[i & (HID - 1)]);
}

// ---------------- WMMA GEMM: out[N,128] = A[N,128](f16) @ W[128,128].T + bias ----------------
// A fragment (16x32 f16): lane half=0 -> K {0..7,16..23}; half=1 -> K {8..15,24..31},
//   VGPR v<4: K = 8*half + 2v; v>=4: K = 16 + 8*half + 2(v-4).
__device__ __forceinline__ v16h load_fragA(const _Float16* __restrict__ base,
                                           int row, int k0, int half) {
  const _Float16* p = base + (size_t)row * HID + k0 + 8 * half;
  v16h f;
#pragma unroll
  for (int q = 0; q < 4; ++q) { f[2 * q] = p[2 * q]; f[2 * q + 1] = p[2 * q + 1]; }
#pragma unroll
  for (int q = 0; q < 4; ++q) { f[8 + 2 * q] = p[16 + 2 * q]; f[9 + 2 * q] = p[16 + 2 * q + 1]; }
  return f;
}

// B fragment (32x16 f16, dense): lane n = lane&15; half=0 -> K {0..7,16..23}, half=1 -> K {8..15,24..31}
// (W row-major [H,128]; B column n == W row n, contiguous in K).
__device__ __forceinline__ v16h load_fragB(const _Float16* __restrict__ base,
                                           int nrow, int k0, int half) {
  const _Float16* p = base + (size_t)nrow * HID + k0 + 8 * half;
  v16h f;
#pragma unroll
  for (int q = 0; q < 8; ++q) { f[2 * q] = p[2 * q + (q >= 4 ? 8 : 0)]; f[2 * q + 1] = p[2 * q + 1 + (q >= 4 ? 8 : 0)]; }
  return f;
}

// Block: 256 threads = 8 waves (one 16-col tile each) x 64 rows (4 M-tiles).
// B fragments loaded once per wave, reused across 4 row tiles -> 16 WMMAs/wave.
__global__ void __launch_bounds__(256)
k_gemm_wmma(const _Float16* __restrict__ A, const _Float16* __restrict__ W16,
            const float* __restrict__ bias, float* __restrict__ out, int Nrows) {
  const int wave  = threadIdx.x >> 5;
  const int lane  = threadIdx.x & 31;
  const int half  = lane >> 4;
  const int l16   = lane & 15;
  const int n0    = wave * 16;          // column tile
  const int mbase = blockIdx.x * 64;    // 4 row tiles per block

  const int brow = n0 + l16;
  v16h bf[4];
#pragma unroll
  for (int ks = 0; ks < 4; ++ks) bf[ks] = load_fragB(W16, brow, ks * 32, half);

  const float bs = bias[n0 + l16];
  const int   n  = n0 + l16;

#pragma unroll
  for (int mt = 0; mt < 4; ++mt) {
    const int m0 = mbase + mt * 16;
    int arow = m0 + l16;
    if (arow >= Nrows) arow = Nrows - 1;   // uniform clamp, no divergence at WMMA

    v8f acc;
#pragma unroll
    for (int q = 0; q < 8; ++q) acc[q] = bs;   // bias folded into C

#pragma unroll
    for (int ks = 0; ks < 4; ++ks) {
      v16h af = load_fragA(A, arow, ks * 32, half);
      acc = __builtin_amdgcn_wmma_f32_16x16x32_f16(
          /*neg_a=*/false, af, /*neg_b=*/false, bf[ks],
          /*c_mod=*/(short)0, acc, /*reuse_a=*/false, /*reuse_b=*/false);
    }

    if (m0 + 16 <= Nrows) {   // scalar (block-uniform) fast path: branch-free stores
#pragma unroll
      for (int q = 0; q < 8; ++q)
        out[(size_t)(m0 + q + 8 * half) * HID + n] = acc[q];
    } else {
#pragma unroll
      for (int q = 0; q < 8; ++q) {
        const int m = m0 + q + 8 * half;
        if (m < Nrows) out[(size_t)m * HID + n] = acc[q];
      }
    }
  }
}

// ---------------- aggregation ----------------

// self-loop: agg = h / deg
__global__ void k_selfinit(const float* __restrict__ h, const float* __restrict__ invdeg,
                           float* __restrict__ agg, long long n) {
  long long i = (long long)blockIdx.x * blockDim.x + threadIdx.x;
  if (i < n) agg[i] = h[i] * invdeg[i >> 7];
}

// edge scatter: agg[c] += norm[e] * h[r]; one wave-lane handles 4 features of one edge
__global__ void k_scatter(const long long* __restrict__ ei, long long E,
                          const float* __restrict__ norm,
                          const float* __restrict__ h, float* __restrict__ agg) {
  long long t = (long long)blockIdx.x * blockDim.x + threadIdx.x;
  long long e = t >> 5;
  if (e >= E) return;
  const int lane = (int)(t & 31);
  const int r = (int)ei[e];
  const int c = (int)ei[E + e];
  const float w = norm[e];
  const float4 v = *(const float4*)(h + (size_t)r * HID + lane * 4);
  float* dst = agg + (size_t)c * HID + lane * 4;
  atomicAdd(dst + 0, w * v.x);
  atomicAdd(dst + 1, w * v.y);
  atomicAdd(dst + 2, w * v.z);
  atomicAdd(dst + 3, w * v.w);
}

// out = relu(agg/deg + bc), written as f16 for the next GEMM
__global__ void k_finalize(const float* __restrict__ agg, const float* __restrict__ invdeg,
                           const float* __restrict__ bc, _Float16* __restrict__ a16,
                           long long n) {
  long long i = (long long)blockIdx.x * blockDim.x + threadIdx.x;
  if (i < n) {
    float v = agg[i] * invdeg[i >> 7] + bc[i & (HID - 1)];
    a16[i] = (_Float16)fmaxf(v, 0.0f);
  }
}

// ---------------- classifier + log_softmax ----------------
// one wave per node; lane j owns classes j and j+32 (C=40)
__global__ void __launch_bounds__(256)
k_classifier(const float* __restrict__ h, const float* __restrict__ W,
             const float* __restrict__ b, float* __restrict__ out, int Nn, int C) {
  __shared__ float sh[8][HID];
  const int wave = threadIdx.x >> 5;
  const int lane = threadIdx.x & 31;
  const int n = blockIdx.x * 8 + wave;
  const int nn = (n < Nn) ? n : (Nn - 1);

#pragma unroll
  for (int i = 0; i < 4; ++i)
    sh[wave][lane * 4 + i] = fmaxf(h[(size_t)nn * HID + lane * 4 + i], 0.0f);  // relu(lin1)
  __syncthreads();

  const int c1 = lane + 32;
  float acc0 = (lane < C) ? b[lane] : -INFINITY;
  float acc1 = (c1 < C) ? b[c1] : -INFINITY;
  for (int k = 0; k < HID; ++k) {
    const float hk = sh[wave][k];
    if (lane < C) acc0 = fmaf(hk, W[(size_t)lane * HID + k], acc0);
    if (c1 < C)   acc1 = fmaf(hk, W[(size_t)c1 * HID + k], acc1);
  }

  float m = fmaxf(acc0, acc1);
#pragma unroll
  for (int off = 16; off > 0; off >>= 1) m = fmaxf(m, __shfl_xor(m, off, 32));
  float s = 0.0f;
  if (lane < C) s += expf(acc0 - m);
  if (c1 < C)   s += expf(acc1 - m);
#pragma unroll
  for (int off = 16; off > 0; off >>= 1) s += __shfl_xor(s, off, 32);
  const float lse = m + logf(s);

  if (n < Nn) {
    if (lane < C) out[(size_t)n * C + lane] = acc0 - lse;
    if (c1 < C)   out[(size_t)n * C + c1]  = acc1 - lse;
  }
}

// ---------------- driver ----------------

extern "C" void kernel_launch(void* const* d_in, const int* in_sizes, int n_in,
                              void* d_out, int out_size, void* d_ws, size_t ws_size,
                              hipStream_t stream) {
  const float*     x      = (const float*)d_in[0];
  const long long* ei     = (const long long*)d_in[1];   // [2,E] int64: row then col
  const float*     imp    = (const float*)d_in[2];
  const float*     Wf[4]  = {(const float*)d_in[3], (const float*)d_in[6],
                             (const float*)d_in[9], (const float*)d_in[12]};
  const float*     bl[4]  = {(const float*)d_in[4], (const float*)d_in[7],
                             (const float*)d_in[10], (const float*)d_in[13]};
  const float*     bc[3]  = {(const float*)d_in[5], (const float*)d_in[8],
                             (const float*)d_in[11]};
  const float*     Wlin2  = (const float*)d_in[14];
  const float*     blin2  = (const float*)d_in[15];

  const int       Nn = in_sizes[0] / HID;
  const long long E  = in_sizes[1] / 2;
  const int       C  = in_sizes[15];

  // workspace bump allocator
  char* p = (char*)d_ws;
  auto bump = [&](size_t bytes) -> void* {
    void* r = (void*)p;
    p += (bytes + 255) & ~(size_t)255;
    return r;
  };
  float*     deg    = (float*)bump((size_t)Nn * 4);
  float*     dinv   = (float*)bump((size_t)Nn * 4);
  float*     invdeg = (float*)bump((size_t)Nn * 4);
  float*     norm   = (float*)bump((size_t)E * 4);
  _Float16*  w16    = (_Float16*)bump((size_t)4 * HID * HID * 2);
  _Float16*  a16    = (_Float16*)bump((size_t)Nn * HID * 2);
  float*     hbuf   = (float*)bump((size_t)Nn * HID * 4);
  float*     agg    = (float*)bump((size_t)Nn * HID * 4);
  (void)ws_size; (void)n_in; (void)out_size;

  const long long NF = (long long)Nn * HID;
  const int  bThr = 256;
  const dim3 blk(bThr);
  const int  gN    = (Nn + bThr - 1) / bThr;
  const int  gE    = (int)((E + bThr - 1) / bThr);
  const int  gNF   = (int)((NF + bThr - 1) / bThr);
  const int  gScat = (int)((E * 32 + bThr - 1) / bThr);
  const int  gemmBlocks = (Nn + 63) / 64;

  // degree / norms
  k_deg_init<<<gN, blk, 0, stream>>>(deg, Nn);
  k_deg_count<<<gE, blk, 0, stream>>>(ei + E, E, deg);
  k_dinv<<<gN, blk, 0, stream>>>(deg, dinv, invdeg, Nn);
  k_norm<<<gE, blk, 0, stream>>>(ei, E, dinv, norm);

  // weights -> f16 (once per call)
  for (int l = 0; l < 4; ++l)
    k_cvt_f16<<<(HID * HID + bThr - 1) / bThr, blk, 0, stream>>>(Wf[l], w16 + (size_t)l * HID * HID, HID * HID);

  // h0 = x * importance (f16 GEMM input)
  k_premul<<<gNF, blk, 0, stream>>>(x, imp, a16, NF);

  // 3 balanced-conv layers
  for (int l = 0; l < 3; ++l) {
    k_gemm_wmma<<<gemmBlocks, blk, 0, stream>>>(a16, w16 + (size_t)l * HID * HID, bl[l], hbuf, Nn);
    k_selfinit<<<gNF, blk, 0, stream>>>(hbuf, invdeg, agg, NF);
    k_scatter<<<gScat, blk, 0, stream>>>(ei, E, norm, hbuf, agg);
    k_finalize<<<gNF, blk, 0, stream>>>(agg, invdeg, bc[l], a16, NF);
  }

  // lin1 (relu applied inside classifier), then classifier + log_softmax
  k_gemm_wmma<<<gemmBlocks, blk, 0, stream>>>(a16, w16 + (size_t)3 * HID * HID, bl[3], hbuf, Nn);
  k_classifier<<<(Nn + 7) / 8, blk, 0, stream>>>(hbuf, Wlin2, blin2, (float*)d_out, Nn, C);
}
